// DecoderModel_73358041415847
// MI455X (gfx1250) — matrix operand order repaired and checked by reading, not compiled, not run on hardware
//
#include <hip/hip_runtime.h>

// ---------------- problem constants ----------------
#define NN 32768
#define GG 1024
#define DD 512
#define HH 768
#define BB 4
#define GHD 64
#define NHD 3
#define GPAD (GG + 64)   // per-branch 16-padding worst case
#define NPAD (NN + 64)

// ---------------- vector types ----------------
typedef __attribute__((ext_vector_type(16))) __bf16 v16bf;
typedef __attribute__((ext_vector_type(8)))  __bf16 v8bf;
typedef __attribute__((ext_vector_type(8)))  float  v8f;

// ---------------- scalar bf16 helpers ----------------
__device__ __forceinline__ __bf16 f2bf(float f) {
    union { float f; unsigned u; } in; in.f = f;
    unsigned r = in.u + 0x7FFFu + ((in.u >> 16) & 1u);   // RNE
    union { unsigned short s; __bf16 b; } out;
    out.s = (unsigned short)(r >> 16);
    return out.b;
}
__device__ __forceinline__ float bf2f(__bf16 b) {
    union { unsigned short s; __bf16 b; } in; in.b = b;
    union { unsigned u; float f; } out;
    out.u = ((unsigned)in.s) << 16;
    return out.f;
}

// ---------------- setup kernels ----------------
__global__ void init_kernel(int* perm_g, int* perm_n, int* cnts) {
    int i = blockIdx.x * blockDim.x + threadIdx.x;
    if (i < GPAD) perm_g[i] = -1;
    if (i < NPAD) perm_n[i] = -1;
    if (i < 16)   cnts[i] = 0;   // gcount[4] ncount[4] gcur[4] ncur[4]
}

// row_start[g] = first node index with batch >= g ; row_start[G] = N
__global__ void row_start_kernel(const int* __restrict__ batch, int* __restrict__ row_start) {
    int n = blockIdx.x * blockDim.x + threadIdx.x;
    if (n >= NN) return;
    int cur = batch[n];
    if (n == 0) {
        for (int g = 0; g <= cur; ++g) row_start[g] = 0;
    } else {
        int prev = batch[n - 1];
        if (cur != prev)
            for (int g = prev + 1; g <= cur; ++g) row_start[g] = n;
    }
    if (n == NN - 1)
        for (int g = cur + 1; g <= GG; ++g) row_start[g] = NN;
}

__global__ void count_kernel(const int* __restrict__ batch, const int* __restrict__ dsname,
                             int* __restrict__ cnts) {
    int i = blockIdx.x * blockDim.x + threadIdx.x;
    if (i < NN) atomicAdd(&cnts[4 + dsname[batch[i]]], 1);
    if (i < GG) atomicAdd(&cnts[dsname[i]], 1);
}

__global__ void bases_kernel(const int* __restrict__ cnts, int* __restrict__ gbase,
                             int* __restrict__ nbase) {
    if (threadIdx.x == 0 && blockIdx.x == 0) {
        int gb = 0, nb = 0;
        for (int b = 0; b < BB; ++b) {
            gbase[b] = gb; nbase[b] = nb;
            gb += (cnts[b]     + 15) & ~15;
            nb += (cnts[4 + b] + 15) & ~15;
        }
        gbase[BB] = gb; nbase[BB] = nb;
    }
}

__global__ void scatter_kernel(const int* __restrict__ batch, const int* __restrict__ dsname,
                               const int* __restrict__ gbase, const int* __restrict__ nbase,
                               int* __restrict__ cur, int* __restrict__ perm_g,
                               int* __restrict__ perm_n) {
    int i = blockIdx.x * blockDim.x + threadIdx.x;
    if (i < GG) {
        int b = dsname[i];
        int pos = gbase[b] + atomicAdd(&cur[b], 1);
        perm_g[pos] = i;
    }
    if (i < NN) {
        int b = dsname[batch[i]];
        int pos = nbase[b] + atomicAdd(&cur[4 + b], 1);
        perm_n[pos] = i;
    }
}

// segment mean: one block per graph
__global__ void segment_mean_kernel(const float* __restrict__ x,
                                    const int* __restrict__ row_start,
                                    float* __restrict__ xg) {
    int g = blockIdx.x;
    int s = row_start[g], e = row_start[g + 1];
    float inv = 1.0f / fmaxf((float)(e - s), 1.0f);
    for (int d = threadIdx.x; d < DD; d += blockDim.x) {
        float sum = 0.0f;
        for (int r = s; r < e; ++r) sum += x[(size_t)r * DD + d];
        xg[(size_t)g * DD + d] = sum * inv;
    }
}

// (B,R,C) f32 -> (B,C,R) bf16  (weight transpose so B-fragments are K-contiguous)
__global__ void transpose_bf16_kernel(const float* __restrict__ src, __bf16* __restrict__ dst,
                                      int R, int C) {
    size_t total = (size_t)BB * R * C;
    size_t stride = (size_t)gridDim.x * blockDim.x;
    for (size_t i = (size_t)blockIdx.x * blockDim.x + threadIdx.x; i < total; i += stride) {
        size_t rc  = (size_t)R * C;
        size_t b   = i / rc;
        size_t rem = i - b * rc;
        size_t c   = rem / R;      // dst linear = b*C*R + c*R + r
        size_t r   = rem - c * R;
        dst[i] = f2bf(src[b * rc + r * C + c]);
    }
}

// ---------------- WMMA fragment loaders ----------------
// A 16x32 bf16 fragment: lane<16 -> K{0..15}, lane>=16 -> K{16..31}
__device__ __forceinline__ v16bf load_a_frag(const __bf16* __restrict__ arow /* &A[M][k0+koff] */) {
    v8bf lo = *(const v8bf*)(arow);
    v8bf hi = *(const v8bf*)(arow + 8);
    return __builtin_shufflevector(lo, hi, 0,1,2,3,4,5,6,7,8,9,10,11,12,13,14,15);
}
// B 32x16 bf16 fragment: lane holds column N=lane%16, 16 contiguous K (kb = lane<16?0:16)
__device__ __forceinline__ v16bf load_b_frag(const __bf16* __restrict__ brow /* &Wt[n][k0+kb] */) {
    v8bf lo = *(const v8bf*)(brow);
    v8bf hi = *(const v8bf*)(brow + 8);
    return __builtin_shufflevector(lo, hi, 0,1,2,3,4,5,6,7,8,9,10,11,12,13,14,15);
}

// ---------------- layer-1 GEMM (graph & node): relu(A_perm @ W_b^T + bias_b) -> bf16 ----------------
// One block = one 16-row M-tile (uniform branch, rows gathered via perm, staged in LDS),
// 8 waves sweep all Nout/16 column tiles. K fixed at 512.
__global__ __launch_bounds__(256) void gemm1_relu_kernel(
    const float* __restrict__ A,        // (rows, 512) f32 source
    const int*   __restrict__ perm,     // padded pos -> src row | -1
    const int*   __restrict__ base,     // [5]
    const __bf16* __restrict__ Wt,      // (B, Nout, 512) bf16 transposed
    const float* __restrict__ bias,     // (B, Nout)
    __bf16* __restrict__ out,           // (Mpad, Nout) bf16
    int Nout) {
    __shared__ __bf16 As[16 * DD];      // 16 KB

    const int tileM = blockIdx.x;
    const int pos0  = tileM * 16;
    const int tid   = threadIdx.x;

    // stage + convert 16 gathered rows of A
    for (int idx = tid; idx < 16 * DD; idx += 256) {
        int r = idx >> 9;               // / 512
        int c = idx & (DD - 1);
        int src = perm[pos0 + r];
        float v = (src >= 0) ? A[(size_t)src * DD + c] : 0.0f;
        As[idx] = f2bf(v);
    }
    __syncthreads();

    int b = 0;
    if (pos0 >= base[1]) b = 1;
    if (pos0 >= base[2]) b = 2;
    if (pos0 >= base[3]) b = 3;

    const int wave  = tid >> 5;
    const int lane  = tid & 31;
    const int half  = lane >> 4;        // 0/1
    const int colN  = lane & 15;
    const int koffA = half ? 16 : 0;
    const int koffB = half ? 16 : 0;

    const __bf16* Wb    = Wt + (size_t)b * Nout * DD;
    const float*  biasb = bias + b * Nout;
    const __bf16* arow0 = &As[colN * DD];

    for (int nt = wave; nt < (Nout >> 4); nt += 8) {
        const int n = nt * 16 + colN;
        const __bf16* wrow = Wb + (size_t)n * DD;
        v8f acc = {};
        #pragma unroll 4
        for (int k0 = 0; k0 < DD; k0 += 32) {
            v16bf af = load_a_frag(arow0 + k0 + koffA);
            v16bf bf = load_b_frag(wrow  + k0 + koffB);
            acc = __builtin_amdgcn_wmma_f32_16x16x32_bf16(
                false, af, false, bf, (short)0, acc, false, false);
        }
        const float bv = biasb[n];
        #pragma unroll
        for (int i = 0; i < 8; ++i) {
            int row = pos0 + i + (half ? 8 : 0);
            float v = acc[i] + bv;
            v = v > 0.0f ? v : 0.0f;
            out[(size_t)row * Nout + n] = f2bf(v);
        }
    }
}

// ---------------- graph layer-2: og = hg @ Wg_head_b^T + bg_head_b ; head/var scatter ----------------
__global__ __launch_bounds__(256) void graph_head_kernel(
    const __bf16* __restrict__ hg,      // (GPAD, 768) bf16
    const int*    __restrict__ perm_g,
    const int*    __restrict__ gbase,
    const __bf16* __restrict__ Wght,    // (B, 128, 768) bf16 transposed
    const float*  __restrict__ bg_head, // (B, 128)
    float* __restrict__ out_head,       // (G, 64)
    float* __restrict__ out_var) {      // (G, 64)
    const int tileM = blockIdx.x;
    const int pos0  = tileM * 16;
    const int wave  = threadIdx.x >> 5;     // 8 waves = 8 N-tiles = 128 cols
    const int lane  = threadIdx.x & 31;
    const int half  = lane >> 4;
    const int colN  = lane & 15;

    int b = 0;
    if (pos0 >= gbase[1]) b = 1;
    if (pos0 >= gbase[2]) b = 2;
    if (pos0 >= gbase[3]) b = 3;

    const int n = wave * 16 + colN;                       // 0..127
    const __bf16* wrow = Wght + ((size_t)b * 128 + n) * HH;
    const __bf16* arow = hg + (size_t)(pos0 + colN) * HH + (half ? 16 : 0);
    const int koffB = half ? 16 : 0;

    v8f acc = {};
    #pragma unroll 4
    for (int k0 = 0; k0 < HH; k0 += 32) {
        v16bf af = load_a_frag(arow + k0);
        v16bf bf = load_b_frag(wrow + k0 + koffB);
        acc = __builtin_amdgcn_wmma_f32_16x16x32_bf16(
            false, af, false, bf, (short)0, acc, false, false);
    }
    const float bv = bg_head[b * 128 + n];
    #pragma unroll
    for (int i = 0; i < 8; ++i) {
        int p = pos0 + i + (half ? 8 : 0);
        int g = perm_g[p];
        if (g >= 0) {
            float v = acc[i] + bv;
            if (n < GHD) out_head[(size_t)g * GHD + n] = v;
            else         out_var [(size_t)g * GHD + (n - GHD)] = v * v;
        }
    }
}

// ---------------- node layer-2: tiny Nout=6, memory-bound VALU + wave32 reduce ----------------
__global__ __launch_bounds__(256) void node_head_kernel(
    const __bf16* __restrict__ hn,      // (NPAD, 512) bf16
    const int*    __restrict__ perm_n,
    const int*    __restrict__ nbase,
    const float*  __restrict__ Wn2,     // (B, 512, 6) f32 (original layout)
    const float*  __restrict__ bn2,     // (B, 6)
    float* __restrict__ out_head,       // (N, 3)
    float* __restrict__ out_var) {      // (N, 3)
    const int p    = blockIdx.x * 8 + (threadIdx.x >> 5);
    const int lane = threadIdx.x & 31;
    if (p >= NPAD) return;
    const int nidx = perm_n[p];
    if (nidx < 0) return;

    int b = 0;
    if (p >= nbase[1]) b = 1;
    if (p >= nbase[2]) b = 2;
    if (p >= nbase[3]) b = 3;

    const float* W = Wn2 + (size_t)b * DD * 6;
    const __bf16* h = hn + (size_t)p * DD;
    float acc[6] = {0, 0, 0, 0, 0, 0};
    for (int k = lane; k < DD; k += 32) {
        float hv = bf2f(h[k]);
        const float* wr = W + k * 6;
        #pragma unroll
        for (int j = 0; j < 6; ++j) acc[j] += hv * wr[j];
    }
    #pragma unroll
    for (int j = 0; j < 6; ++j)
        for (int off = 16; off > 0; off >>= 1)
            acc[j] += __shfl_xor(acc[j], off, 32);
    if (lane == 0) {
        #pragma unroll
        for (int j = 0; j < NHD; ++j)
            out_head[(size_t)nidx * NHD + j] = acc[j] + bn2[b * 6 + j];
        #pragma unroll
        for (int j = NHD; j < 6; ++j) {
            float v = acc[j] + bn2[b * 6 + j];
            out_var[(size_t)nidx * NHD + (j - NHD)] = v * v;
        }
    }
}

// ---------------- host-side launcher ----------------
extern "C" void kernel_launch(void* const* d_in, const int* in_sizes, int n_in,
                              void* d_out, int out_size, void* d_ws, size_t ws_size,
                              hipStream_t stream) {
    const float* x         = (const float*)d_in[0];
    const int*   batch     = (const int*)  d_in[1];
    const int*   dsname    = (const int*)  d_in[2];
    const float* Wg_shared = (const float*)d_in[3];
    const float* bg_shared = (const float*)d_in[4];
    const float* Wg_head   = (const float*)d_in[5];
    const float* bg_head   = (const float*)d_in[6];
    const float* Wn1       = (const float*)d_in[7];
    const float* bn1       = (const float*)d_in[8];
    const float* Wn2       = (const float*)d_in[9];
    const float* bn2       = (const float*)d_in[10];
    float* out = (float*)d_out;

    float* out_gh = out;                                  // (G, 64)
    float* out_nh = out + (size_t)GG * GHD;               // (N, 3)
    float* out_gv = out_nh + (size_t)NN * NHD;            // (G, 64)
    float* out_nv = out_gv + (size_t)GG * GHD;            // (N, 3)

    // workspace carve-up (all 256B aligned)
    char* ws = (char*)d_ws;
    size_t off = 0;
    auto carve = [&](size_t bytes) -> char* {
        char* p = ws + off;
        off = (off + bytes + 255) & ~(size_t)255;
        return p;
    };
    int*    row_start = (int*)   carve((GG + 1) * sizeof(int));
    int*    cnts      = (int*)   carve(16 * sizeof(int));       // gcount4 ncount4 gcur4 ncur4
    int*    gbase     = (int*)   carve(8 * sizeof(int));
    int*    nbase     = (int*)   carve(8 * sizeof(int));
    int*    perm_g    = (int*)   carve(GPAD * sizeof(int));
    int*    perm_n    = (int*)   carve(NPAD * sizeof(int));
    float*  xg        = (float*) carve((size_t)GG * DD * sizeof(float));
    __bf16* Wg_t      = (__bf16*)carve((size_t)BB * HH * DD * sizeof(__bf16));
    __bf16* Wn1_t     = (__bf16*)carve((size_t)BB * DD * DD * sizeof(__bf16));
    __bf16* Wgh_t     = (__bf16*)carve((size_t)BB * 128 * HH * sizeof(__bf16));
    __bf16* hg        = (__bf16*)carve((size_t)GPAD * HH * sizeof(__bf16));
    __bf16* hn        = (__bf16*)carve((size_t)NPAD * DD * sizeof(__bf16));
    (void)ws_size; (void)in_sizes; (void)n_in; (void)out_size;

    const int T = 256;
    // 1) init perms/counters
    init_kernel<<<(NPAD + T - 1) / T, T, 0, stream>>>(perm_g, perm_n, cnts);
    // 2) CSR row starts from sorted batch
    row_start_kernel<<<(NN + T - 1) / T, T, 0, stream>>>(batch, row_start);
    // 3) branch histograms
    count_kernel<<<(NN + T - 1) / T, T, 0, stream>>>(batch, dsname, cnts);
    // 4) 16-aligned branch bases
    bases_kernel<<<1, 64, 0, stream>>>(cnts, gbase, nbase);
    // 5) counting-sort scatter (branch-uniform 16-row tiles)
    scatter_kernel<<<(NN + T - 1) / T, T, 0, stream>>>(batch, dsname, gbase, nbase,
                                                       cnts + 8, perm_g, perm_n);
    // 6) segment mean
    segment_mean_kernel<<<GG, 128, 0, stream>>>(x, row_start, xg);
    // 7) weight transpose + bf16 convert
    transpose_bf16_kernel<<<2048, T, 0, stream>>>(Wg_shared, Wg_t, DD, HH);
    transpose_bf16_kernel<<<2048, T, 0, stream>>>(Wn1, Wn1_t, DD, DD);
    transpose_bf16_kernel<<<1024, T, 0, stream>>>(Wg_head, Wgh_t, HH, 2 * GHD);
    // 8) graph layer 1: (GPAD x 512) @ (512 x 768) -> relu -> hg
    gemm1_relu_kernel<<<GPAD / 16, T, 0, stream>>>(xg, perm_g, gbase, Wg_t, bg_shared, hg, HH);
    // 9) node layer 1 (dominant GEMM): (NPAD x 512) @ (512 x 512) -> relu -> hn
    gemm1_relu_kernel<<<NPAD / 16, T, 0, stream>>>(x, perm_n, nbase, Wn1_t, bn1, hn, DD);
    // 10) graph layer 2 + head/var scatter
    graph_head_kernel<<<GPAD / 16, T, 0, stream>>>(hg, perm_g, gbase, Wgh_t, bg_head,
                                                   out_gh, out_gv);
    // 11) node layer 2 (Nout=6) + head/var scatter
    node_head_kernel<<<NPAD / 8, T, 0, stream>>>(hn, perm_n, nbase, Wn2, bn2,
                                                 out_nh, out_nv);
}
